// LanguageModel_12962211300063
// MI455X (gfx1250) — compile-verified, matrix operand order checked by reading, not run on hardware
//
#include <hip/hip_runtime.h>
#include <hip/hip_bf16.h>

// ---------------------------------------------------------------------------
// LSTM language model on gfx1250 (MI455X), bf16 WMMA everywhere.
// V=20000, S=30 (T=29), E=100, H=1024, O=512, B=64
// Recurrence parallelized over 32 persistent blocks (grid barrier) with
// async-to-LDS staging of the h state each step.
// ---------------------------------------------------------------------------

typedef __attribute__((ext_vector_type(16))) __bf16 v16bf;
typedef __attribute__((ext_vector_type(8)))  float  v8f;

#define WMMA_BF16(a, b, c) \
  __builtin_amdgcn_wmma_f32_16x16x32_bf16(false, (a), false, (b), (short)0, (c), false, false)

// K index inside a 32-wide K-tile for 16-bit A/B fragments (ISA 7.12.2):
// VGPR v (0..7) holds bf16 pair (bit 0/1); lanes 0-15 vs 16-31 offset K by 8.
static __device__ __forceinline__ int kk_of(int v, int bit, int half) {
  return (v < 4 ? 2 * v : 16 + 2 * (v - 4)) + bit + half * 8;
}

static __device__ __forceinline__ float hsig(float x) {
  return fminf(fmaxf(0.2f * x + 0.5f, 0.0f), 1.0f);
}

union AFrag { v16bf v; unsigned u[8]; };

// Monotonic centralized grid barrier (k-th invocation waits for nb*k arrivals).
static __device__ __forceinline__ void grid_barrier(int* cnt, int nb, int k) {
  __threadfence();
  __syncthreads();
  if (threadIdx.x == 0) {
    __hip_atomic_fetch_add(cnt, 1, __ATOMIC_RELEASE, __HIP_MEMORY_SCOPE_AGENT);
    while (__hip_atomic_load(cnt, __ATOMIC_ACQUIRE, __HIP_MEMORY_SCOPE_AGENT) < nb * k)
      __builtin_amdgcn_s_sleep(1);
  }
  __syncthreads();
  __threadfence();
}

// ---------------------------------------------------------------------------
// Pack a row-major f32 [K x N] matrix into bf16 B-fragments:
// dst[((kt*NT + nt)*32 + lane)*16 + e], zero-padded outside K x N.
// ---------------------------------------------------------------------------
__global__ void pack_b_kernel(const float* __restrict__ src, __bf16* __restrict__ dst,
                              int K, int N, int KT, int NT) {
  long long total = (long long)KT * NT * 512;
  for (long long p = (long long)blockIdx.x * blockDim.x + threadIdx.x; p < total;
       p += (long long)gridDim.x * blockDim.x) {
    int e    = (int)(p & 15);
    int lane = (int)((p >> 4) & 31);
    long long frag = p >> 9;
    int nt = (int)(frag % NT);
    int kt = (int)(frag / NT);
    int v = e >> 1, bit = e & 1, half = lane >> 4;
    int n = nt * 16 + (lane & 15);
    int k = kt * 32 + kk_of(v, bit, half);
    float val = (k < K && n < N) ? src[(long long)k * N + n] : 0.0f;
    dst[p] = (__bf16)val;
  }
}

// ---------------------------------------------------------------------------
// Gather embeddings for all (t,b) rows and pack as bf16 A-fragments:
// rows R = t*64 + b (R=0..1855 -> 116 mtiles), K = E padded 100 -> 128 (4 kt).
// ---------------------------------------------------------------------------
__global__ void pack_x_kernel(const int* __restrict__ ids, const float* __restrict__ emb,
                              __bf16* __restrict__ dst) {
  const long long total = 116LL * 4 * 512;
  for (long long p = (long long)blockIdx.x * blockDim.x + threadIdx.x; p < total;
       p += (long long)gridDim.x * blockDim.x) {
    int e    = (int)(p & 15);
    int lane = (int)((p >> 4) & 31);
    long long frag = p >> 9;
    int kt = (int)(frag & 3);
    int rt = (int)(frag >> 2);
    int v = e >> 1, bit = e & 1, half = lane >> 4;
    int m = rt * 16 + (lane & 15);      // global row R = t*64 + b
    int t = m >> 6;
    int b = m & 63;
    int k = kt * 32 + kk_of(v, bit, half);
    float val = 0.0f;
    if (k < 100) {
      int tok = ids[b * 30 + t];        // sentence_ids[b, t]
      val = emb[(long long)tok * 100 + k];
    }
    dst[p] = (__bf16)val;
  }
}

// ---------------------------------------------------------------------------
// Zero h0 (packed), c0, and the grid-barrier counter.
// ---------------------------------------------------------------------------
__global__ void init_state_kernel(unsigned* __restrict__ hpack32,
                                  float* __restrict__ cstate, int* __restrict__ cnt) {
  int g = blockIdx.x * blockDim.x + threadIdx.x;
  if (g == 0) *cnt = 0;
  if (g < 32768) hpack32[g] = 0u;       // 65536 bf16
  if (g < 65536) cstate[g] = 0.0f;
}

// ---------------------------------------------------------------------------
// Phase 1: xz[R=1856, G=4096] = x @ kernel + bias  (one WMMA tile per wave)
// ---------------------------------------------------------------------------
__global__ __launch_bounds__(256) void gemm_xz_kernel(const __bf16* __restrict__ xpack,
                                                      const __bf16* __restrict__ kpack,
                                                      const float* __restrict__ bias,
                                                      float* __restrict__ xz) {
  const int NT = 256, KT = 4;
  long long tile = (long long)blockIdx.x * 8 + (threadIdx.x >> 5);
  if (tile >= 116LL * NT) return;       // wave-uniform
  int mt = (int)(tile / NT);
  int nt = (int)(tile % NT);
  int lane = threadIdx.x & 31;
  v8f acc = {};
#pragma unroll
  for (int kt = 0; kt < KT; ++kt) {
    v16bf a = *(const v16bf*)(xpack + (((long long)mt * KT + kt) * 32 + lane) * 16);
    v16bf b = *(const v16bf*)(kpack + (((long long)kt * NT + nt) * 32 + lane) * 16);
    acc = WMMA_BF16(a, b, acc);
  }
  int col  = nt * 16 + (lane & 15);
  int half = lane >> 4;
  float bv = bias[col];
#pragma unroll
  for (int r = 0; r < 8; ++r) {
    int row = mt * 16 + r + 8 * half;
    xz[(long long)row * 4096 + col] = acc[r] + bv;
  }
}

// ---------------------------------------------------------------------------
// Phase 2: sequential LSTM over 29 steps, 32 persistent blocks x 8 waves.
// Packed h state lives in global ws (L2); each block async-copies it into
// its 128 KB LDS every step (global_load_async_to_lds_b128, ASYNCcnt).
// Per step: [zrec GEMM (4 tiles/wave) + projection of previous step]
//           -> grid barrier -> gates/state update -> grid barrier.
// ---------------------------------------------------------------------------
__global__ __launch_bounds__(256) void lstm_seq_kernel(
    const float* __restrict__ xz, const __bf16* __restrict__ recpack,
    const __bf16* __restrict__ wppack, const float* __restrict__ bp,
    __bf16* __restrict__ hpack, float* __restrict__ zrec,
    float* __restrict__ cstate, __bf16* __restrict__ outbuf, int* cnt) {
  extern __shared__ __bf16 hlds[];                  // 65536 bf16 = 128 KB
  const int tid  = threadIdx.x;
  const int lane = tid & 31;
  const int wg   = blockIdx.x * 8 + (tid >> 5);     // global wave id 0..255
  const int g    = blockIdx.x * 256 + tid;          // global thread id 0..8191
  const int NB   = 32;
  int bk = 0;

  const unsigned ldsbase = (unsigned)(unsigned long long)(void*)hlds;

  for (int t = 0; t < 29; ++t) {
    // ---- stage h_t (packed) into LDS via async copy ----
    {
      const char* src = (const char*)hpack;
#pragma unroll 4
      for (int i = 0; i < 32; ++i) {
        unsigned off = (unsigned)(tid * 16 + i * 4096);
        asm volatile("global_load_async_to_lds_b128 %0, %1, off"
                     :: "v"(ldsbase + off), "v"(src + off) : "memory");
      }
      asm volatile("s_wait_asynccnt 0" ::: "memory");
      __syncthreads();
    }

    // ---- zrec = h_t @ rec_kernel : 1024 tiles, 4 per wave ----
#pragma unroll
    for (int i = 0; i < 4; ++i) {
      int tile = wg + i * 256;
      int mt = tile >> 8;
      int nt = tile & 255;
      v8f acc = {};
#pragma unroll 4
      for (int kt = 0; kt < 32; ++kt) {
        v16bf a = *(const v16bf*)(hlds + ((mt * 32 + kt) * 32 + lane) * 16);
        v16bf b = *(const v16bf*)(recpack + (((long long)kt * 256 + nt) * 32 + lane) * 16);
        acc = WMMA_BF16(a, b, acc);
      }
      int col  = nt * 16 + (lane & 15);
      int half = lane >> 4;
#pragma unroll
      for (int r = 0; r < 8; ++r)
        zrec[(long long)(mt * 16 + r + 8 * half) * 4096 + col] = acc[r];
    }

    // ---- projection for step t-1: out_{t-1} = h_t @ Wp + bp (128 tiles) ----
    if (t > 0 && wg < 128) {
      int mt = wg >> 5;
      int nt = wg & 31;
      v8f acc = {};
#pragma unroll 4
      for (int kt = 0; kt < 32; ++kt) {
        v16bf a = *(const v16bf*)(hlds + ((mt * 32 + kt) * 32 + lane) * 16);
        v16bf b = *(const v16bf*)(wppack + (((long long)kt * 32 + nt) * 32 + lane) * 16);
        acc = WMMA_BF16(a, b, acc);
      }
      int col  = nt * 16 + (lane & 15);
      int half = lane >> 4;
      float bv = bp[col];
#pragma unroll
      for (int r = 0; r < 8; ++r) {
        int m = mt * 16 + r + 8 * half;
        outbuf[((long long)((t - 1) * 64 + m)) * 512 + col] = (__bf16)(acc[r] + bv);
      }
    }

    grid_barrier(cnt, NB, ++bk);

    // ---- gates + state update: 65536 elems, 8 per thread ----
    const float* xzt = xz + (long long)t * 64 * 4096;
#pragma unroll
    for (int j = 0; j < 8; ++j) {
      int q = g + j * 8192;
      int m = q >> 10;
      int h = q & 1023;
      long long base = (long long)m * 4096 + h;
      float zi = xzt[base]        + zrec[base];
      float zf = xzt[base + 1024] + zrec[base + 1024];
      float zc = xzt[base + 2048] + zrec[base + 2048];
      float zo = xzt[base + 3072] + zrec[base + 3072];
      float ig = hsig(zi), fg = hsig(zf), og = hsig(zo);
      float c  = fg * cstate[q] + ig * tanhf(zc);
      cstate[q] = c;
      float hn = og * tanhf(c);
      // scatter h_{t+1} into fragment-packed global layout (inverse of kk_of)
      int kt = h >> 5, kk = h & 31;
      int up = kk >> 4, rem = kk & 15, hf = rem >> 3, r2 = rem & 7;
      int v = up * 4 + (r2 >> 1), bit = r2 & 1;
      int e  = 2 * v + bit;
      int ln = (m & 15) + (hf << 4);
      int mt = m >> 4;
      hpack[((mt * 32 + kt) * 32 + ln) * 16 + e] = (__bf16)hn;
    }

    grid_barrier(cnt, NB, ++bk);
  }

  // ---- epilogue: projection for t=28 using final h_29 ----
  if (blockIdx.x < 16) {
    const char* src = (const char*)hpack;
#pragma unroll 4
    for (int i = 0; i < 32; ++i) {
      unsigned off = (unsigned)(tid * 16 + i * 4096);
      asm volatile("global_load_async_to_lds_b128 %0, %1, off"
                   :: "v"(ldsbase + off), "v"(src + off) : "memory");
    }
    asm volatile("s_wait_asynccnt 0" ::: "memory");
    __syncthreads();
    int mt = wg >> 5;
    int nt = wg & 31;
    v8f acc = {};
#pragma unroll 4
    for (int kt = 0; kt < 32; ++kt) {
      v16bf a = *(const v16bf*)(hlds + ((mt * 32 + kt) * 32 + lane) * 16);
      v16bf b = *(const v16bf*)(wppack + (((long long)kt * 32 + nt) * 32 + lane) * 16);
      acc = WMMA_BF16(a, b, acc);
    }
    int col  = nt * 16 + (lane & 15);
    int half = lane >> 4;
    float bv = bp[col];
#pragma unroll
    for (int r = 0; r < 8; ++r) {
      int m = mt * 16 + r + 8 * half;
      outbuf[((long long)(28 * 64 + m)) * 512 + col] = (__bf16)(acc[r] + bv);
    }
  }
}

// ---------------------------------------------------------------------------
// Phase 3: logits[1856, 20000] = out @ W. Block = 8 waves, one mtile per
// block (A staged in LDS), one ntile per wave. D written straight to d_out
// with the [B, T, V] remap.
// ---------------------------------------------------------------------------
__global__ __launch_bounds__(256) void gemm_logits_kernel(const __bf16* __restrict__ outbuf,
                                                          const __bf16* __restrict__ wpack,
                                                          float* __restrict__ logits) {
  __shared__ __bf16 atile[16 * 512];            // 16 KB: 16 rows x K=512 bf16
  const int mt = blockIdx.y;                    // 0..115
  const int tid = threadIdx.x;
  const int lane = tid & 31;
  const int wave = tid >> 5;

  const unsigned* src32 = (const unsigned*)(outbuf + (long long)mt * 16 * 512);
  unsigned* dst32 = (unsigned*)atile;
  for (int i = tid; i < 4096; i += 256) dst32[i] = src32[i];
  __syncthreads();

  int nt = blockIdx.x * 8 + wave;
  if (nt >= 1250) return;                       // wave-uniform

  const int mrow = lane & 15;
  const int half = lane >> 4;
  v8f acc = {};
  for (int kt = 0; kt < 16; ++kt) {
    if (kt + 1 < 16)
      __builtin_prefetch(wpack + (((long long)(kt + 1) * 1250 + nt) * 32 + lane) * 16, 0, 1);
    AFrag a;
#pragma unroll
    for (int v = 0; v < 8; ++v) {               // bf16 K-pairs are dword-adjacent
      int k = kt * 32 + (v < 4 ? 2 * v : 16 + 2 * (v - 4)) + half * 8;
      a.u[v] = *(const unsigned*)(atile + mrow * 512 + k);
    }
    v16bf b = *(const v16bf*)(wpack + (((long long)kt * 1250 + nt) * 32 + lane) * 16);
    acc = WMMA_BF16(a.v, b, acc);
  }
  int col = nt * 16 + (lane & 15);
#pragma unroll
  for (int r = 0; r < 8; ++r) {
    int row = mt * 16 + r + 8 * half;           // R = t*64 + b
    int t = row >> 6;
    int b = row & 63;
    logits[((long long)(b * 29 + t)) * 20000 + col] = acc[r];
  }
}

// ---------------------------------------------------------------------------
extern "C" void kernel_launch(void* const* d_in, const int* in_sizes, int n_in,
                              void* d_out, int out_size, void* d_ws, size_t ws_size,
                              hipStream_t stream) {
  const int*   ids  = (const int*)d_in[0];
  const float* emb  = (const float*)d_in[1];
  const float* ker  = (const float*)d_in[2];
  const float* rec  = (const float*)d_in[3];
  const float* bias = (const float*)d_in[4];
  const float* Wp   = (const float*)d_in[5];
  const float* bp   = (const float*)d_in[6];
  const float* W    = (const float*)d_in[7];
  float* logits = (float*)d_out;

  char* ws = (char*)d_ws;
  size_t off = 0;
  auto take = [&](size_t bytes) {
    size_t cur = off;
    off = (off + bytes + 255) & ~(size_t)255;
    return (void*)(ws + cur);
  };
  __bf16* xpack   = (__bf16*)take(116LL * 4 * 512 * 2);        // x A-fragments
  __bf16* kpack   = (__bf16*)take(4LL * 256 * 512 * 2);        // kernel
  __bf16* recpack = (__bf16*)take(32LL * 256 * 512 * 2);       // rec_kernel
  __bf16* wppack  = (__bf16*)take(32LL * 32 * 512 * 2);        // Wp
  __bf16* wpack   = (__bf16*)take(16LL * 1250 * 512 * 2);      // W
  float*  xz      = (float*)take(29LL * 64 * 4096 * 4);        // xz = x@kernel+bias
  float*  zrec    = (float*)take(64LL * 4096 * 4);             // per-step h@rec
  float*  cstate  = (float*)take(64LL * 1024 * 4);             // cell state
  __bf16* hpack   = (__bf16*)take(65536LL * 2);                // packed h state
  __bf16* outbuf  = (__bf16*)take(1856LL * 512 * 2);           // projected h (A side)
  int*    barcnt  = (int*)take(256);                           // grid barrier counter

  init_state_kernel<<<256, 256, 0, stream>>>((unsigned*)hpack, cstate, barcnt);

  pack_x_kernel<<<232, 256, 0, stream>>>(ids, emb, xpack);
  pack_b_kernel<<<1024, 256, 0, stream>>>(ker, kpack, 100, 4096, 4, 256);
  pack_b_kernel<<<4096, 256, 0, stream>>>(rec, recpack, 1024, 4096, 32, 256);
  pack_b_kernel<<<1024, 256, 0, stream>>>(Wp, wppack, 1024, 512, 32, 32);
  pack_b_kernel<<<8192, 256, 0, stream>>>(W, wpack, 512, 20000, 16, 1250);

  gemm_xz_kernel<<<3712, 256, 0, stream>>>(xpack, kpack, bias, xz);

  lstm_seq_kernel<<<32, 256, 131072, stream>>>(xz, recpack, wppack, bp,
                                               hpack, zrec, cstate, outbuf, barcnt);

  gemm_logits_kernel<<<dim3(157, 116), 256, 0, stream>>>(outbuf, wpack, logits);
}
